// GraphAttention_54331336294749
// MI455X (gfx1250) — matrix operand order repaired and checked by reading, not compiled, not run on hardware
//
#include <hip/hip_runtime.h>

typedef __bf16 bf16;
typedef __attribute__((ext_vector_type(8)))  __bf16 v8bf;
typedef __attribute__((ext_vector_type(16))) __bf16 v16bf;
typedef __attribute__((ext_vector_type(8)))  float  v8f;
typedef __attribute__((ext_vector_type(4)))  float  v4f;

#define ALPHA 0.1f
static constexpr int N    = 8192;   // rows of feat_edge / output
static constexpr int NA   = 8192;   // rows of feat_edge_a / feat_node_a
static constexpr int EF   = 256;    // edge features
static constexpr int INF_ = 512;    // in_features
static constexpr int OUTF = 512;    // out_features

// ---- workspace layout (bytes) ----
static constexpr size_t OFF_S1   = 0;                                   // 32KB f32
static constexpr size_t OFF_S2   = 32768;                               // 32KB f32
static constexpr size_t OFF_M    = 65536;                               // 32KB f32 row max
static constexpr size_t OFF_IZ   = 98304;                               // 32KB f32 1/Z
static constexpr size_t OFF_MAX  = 131072;                              // 4B   f32 max(s2)
static constexpr size_t OFF_XBF  = (size_t)1 << 20;                     // 8MB  bf16 X
static constexpr size_t OFF_WT   = OFF_XBF + (size_t)NA * INF_ * 2;     // 512KB bf16 W^T
static constexpr size_t OFF_WHT  = OFF_WT + (size_t)OUTF * INF_ * 2;    // 8MB  bf16 Wh^T

__device__ __forceinline__ v16bf cat8(v8bf lo, v8bf hi) {
  v16bf r;
#pragma unroll
  for (int i = 0; i < 8; ++i) { r[i] = lo[i]; r[i + 8] = hi[i]; }
  return r;
}

// ---------------- f32 -> bf16 convert (feat_node_a) ----------------
__global__ __launch_bounds__(256) void cvt_x_kernel(const float* __restrict__ x,
                                                    bf16* __restrict__ o, int n) {
  int i = (blockIdx.x * 256 + threadIdx.x) * 4;
  if (i + 3 < n) {
    v4f v = *(const v4f*)(x + i);
    o[i + 0] = (bf16)v[0]; o[i + 1] = (bf16)v[1];
    o[i + 2] = (bf16)v[2]; o[i + 3] = (bf16)v[3];
  }
}

// ---------------- weight [K][Nout] f32 -> W^T [Nout][K] bf16 ----------------
__global__ __launch_bounds__(256) void cvt_wT_kernel(const float* __restrict__ w,
                                                     bf16* __restrict__ wt) {
  int idx = blockIdx.x * 256 + threadIdx.x;      // 0 .. 512*512-1
  int c = idx >> 9, k = idx & 511;
  wt[(size_t)c * INF_ + k] = (bf16)w[(size_t)k * OUTF + c];
}

// ---------------- s1 = feat_edge @ att_lo, s2 = feat_edge_a @ att_hi ----------------
__global__ __launch_bounds__(256) void s_kernel(const float* __restrict__ fe,
                                                const float* __restrict__ fea,
                                                const float* __restrict__ att,
                                                float* __restrict__ s1,
                                                float* __restrict__ s2) {
  int wave = threadIdx.x >> 5, lane = threadIdx.x & 31;
  int g = blockIdx.x * 8 + wave;                 // 0 .. 16383 : one wave per row
  const float* src; const float* a; float* dst;
  if (g < N) { src = fe  + (size_t)g * EF;       a = att;      dst = s1 + g; }
  else       { src = fea + (size_t)(g - N) * EF; a = att + EF; dst = s2 + (g - N); }
  float sum = 0.f;
#pragma unroll
  for (int t = 0; t < EF / 32; ++t) {
    int i = lane + 32 * t;
    sum += src[i] * a[i];
  }
#pragma unroll
  for (int off = 16; off > 0; off >>= 1) sum += __shfl_xor(sum, off, 32);
  if (lane == 0) *dst = sum;
}

// ---------------- s2max = max_j s2[j] ----------------
__global__ __launch_bounds__(256) void max_kernel(const float* __restrict__ s2,
                                                  float* __restrict__ s2max) {
  __shared__ float red[256];
  int tid = threadIdx.x;
  float mx = -3.4e38f;
  for (int i = tid; i < NA; i += 256) mx = fmaxf(mx, s2[i]);
  red[tid] = mx; __syncthreads();
  for (int o = 128; o > 0; o >>= 1) {
    if (tid < o) red[tid] = fmaxf(red[tid], red[tid + o]);
    __syncthreads();
  }
  if (tid == 0) *s2max = red[0];
}

// ---------------- per-row softmax stats: m_i, 1/Z_i ----------------
__global__ __launch_bounds__(256) void z_kernel(const float* __restrict__ s1,
                                                const float* __restrict__ s2,
                                                const float* __restrict__ s2max,
                                                float* __restrict__ mrow,
                                                float* __restrict__ invZ) {
  __shared__ float red[256];
  int i = blockIdx.x, tid = threadIdx.x;
  float s1i = s1[i];
  float xm = s1i + *s2max;
  float mi = xm > 0.f ? xm : ALPHA * xm;         // lrelu monotone -> row max
  float sum = 0.f;
  for (int j = tid; j < NA; j += 256) {
    float y = s1i + s2[j];
    float e = y > 0.f ? y : ALPHA * y;
    sum += __expf(e - mi);
  }
  red[tid] = sum; __syncthreads();
  for (int o = 128; o > 0; o >>= 1) {
    if (tid < o) red[tid] += red[tid + o];
    __syncthreads();
  }
  if (tid == 0) { mrow[i] = mi; invZ[i] = 1.f / red[0]; }
}

// ---------------- Wh^T[c][j] = (X @ W)[j][c] in bf16, via bf16 WMMA ----------------
__global__ __launch_bounds__(256) void wh_kernel(const bf16* __restrict__ X,
                                                 const bf16* __restrict__ WT,
                                                 bf16* __restrict__ whT) {
  int wave = threadIdx.x >> 5, lane = threadIdx.x & 31;
  int idx = blockIdx.x * 8 + wave;               // 0 .. 16383 : one 16x16 tile per wave
  int tm = idx >> 5;                             // 0..511 (rows of X)
  int tn = idx & 31;                             // 0..31  (output cols)
  int m0 = tm * 16, n0 = tn * 16;
  int lr   = lane & 15;
  int koff = (lane < 16) ? 0 : 16;               // A-fragment contiguous-K offset
  int boff = (lane < 16) ? 0 : 16;               // B-fragment contiguous-K offset
  const bf16* arow = X  + (size_t)(m0 + lr) * INF_;
  const bf16* brow = WT + (size_t)(n0 + lr) * INF_;
  v8f acc = {};
  for (int k0 = 0; k0 < INF_; k0 += 32) {
    v16bf a  = *(const v16bf*)(arow + k0 + koff);
    v16bf b  = *(const v16bf*)(brow + k0 + boff);
    acc = __builtin_amdgcn_wmma_f32_16x16x32_bf16(false, a, false, b,
                                                  (short)0, acc, false, false);
  }
  // transposed store: contiguous 8 bf16 along the j (row) axis of Wh^T
  int row0 = m0 + ((lane < 16) ? 0 : 8);
  int col  = n0 + lr;
  v8bf o;
#pragma unroll
  for (int r = 0; r < 8; ++r) o[r] = (bf16)acc[r];
  *(v8bf*)(whT + (size_t)col * NA + row0) = o;
}

// ---------------- fused softmax(rank-1) @ Wh, + 1/Z scale + ELU ----------------
// 512 threads = 16 waves; block handles 64 rows x all 512 output cols.
// wave w: row tile = w>>2, col group = (w&3)*128 (8 x 16x16 C tiles per wave).
__global__ __launch_bounds__(512) void attn_kernel(const float* __restrict__ s1,
                                                   const float* __restrict__ s2,
                                                   const float* __restrict__ mrow,
                                                   const float* __restrict__ invZ,
                                                   const bf16* __restrict__ whT,
                                                   float* __restrict__ out) {
  __shared__ __align__(16) bf16 P[64 * 32];      // 4KB probability tile
  int tid  = threadIdx.x;
  int wave = tid >> 5, lane = tid & 31;
  int block0 = blockIdx.x * 64;
  int rt = wave >> 2;
  int n0 = (wave & 3) * 128;
  int lr   = lane & 15;
  int koff = (lane < 16) ? 0 : 16;
  int boff = (lane < 16) ? 0 : 16;
  int arow = rt * 16 + lr;                       // row within block for A fragment

  // cooperative P computation: each thread owns 4 (row,k) entries
  int pr = tid >> 3;                             // 0..63
  int pk = (tid & 7) * 4;                        // 0,4,..,28
  float s1r = s1[block0 + pr];
  float mr  = mrow[block0 + pr];

  v8f acc[8] = {};

  for (int j0 = 0; j0 < NA; j0 += 32) {
    __syncthreads();                             // P consumed by previous iter
    v4f sv = *(const v4f*)(s2 + j0 + pk);
#pragma unroll
    for (int q = 0; q < 4; ++q) {
      float x = s1r + sv[q];
      float e = x > 0.f ? x : ALPHA * x;
      P[pr * 32 + pk + q] = (bf16)__expf(e - mr);
    }
    __syncthreads();
    v16bf a  = *(const v16bf*)(&P[arow * 32 + koff]);
#pragma unroll
    for (int t = 0; t < 8; ++t) {
      const bf16* bp = whT + (size_t)(n0 + t * 16 + lr) * NA + j0 + boff;
      v16bf b = *(const v16bf*)bp;
      acc[t] = __builtin_amdgcn_wmma_f32_16x16x32_bf16(false, a, false, b,
                                                       (short)0, acc[t], false, false);
    }
  }

  // epilogue: C tile layout — VGPR r: M = r (+8 for lanes 16..31), N = lane&15
  int mbase = block0 + rt * 16 + ((lane < 16) ? 0 : 8);
  float iz[8];
#pragma unroll
  for (int r = 0; r < 8; ++r) iz[r] = invZ[mbase + r];
#pragma unroll
  for (int t = 0; t < 8; ++t) {
    int col = n0 + t * 16 + lr;
#pragma unroll
    for (int r = 0; r < 8; ++r) {
      float v = acc[t][r] * iz[r];
      v = v > 0.f ? v : (__expf(v) - 1.f);       // ELU
      out[(size_t)(mbase + r) * OUTF + col] = v;
    }
  }
}

extern "C" void kernel_launch(void* const* d_in, const int* in_sizes, int n_in,
                              void* d_out, int out_size, void* d_ws, size_t ws_size,
                              hipStream_t stream) {
  const float* feat_edge   = (const float*)d_in[0];
  const float* feat_edge_a = (const float*)d_in[1];
  const float* feat_node_a = (const float*)d_in[2];
  const float* weight      = (const float*)d_in[3];
  const float* att         = (const float*)d_in[4];
  float* out = (float*)d_out;

  char* ws = (char*)d_ws;
  float* s1    = (float*)(ws + OFF_S1);
  float* s2    = (float*)(ws + OFF_S2);
  float* mrow  = (float*)(ws + OFF_M);
  float* invZ  = (float*)(ws + OFF_IZ);
  float* s2max = (float*)(ws + OFF_MAX);
  bf16*  Xbf   = (bf16*)(ws + OFF_XBF);
  bf16*  WTbf  = (bf16*)(ws + OFF_WT);
  bf16*  whT   = (bf16*)(ws + OFF_WHT);

  // 1) precision conversion
  cvt_x_kernel <<<(NA * INF_) / (256 * 4), 256, 0, stream>>>(feat_node_a, Xbf, NA * INF_);
  cvt_wT_kernel<<<(INF_ * OUTF) / 256,     256, 0, stream>>>(weight, WTbf);
  // 2) rank-1 attention score vectors
  s_kernel<<<(N + NA) / 8, 256, 0, stream>>>(feat_edge, feat_edge_a, att, s1, s2);
  // 3) global max of s2 -> closed-form row max
  max_kernel<<<1, 256, 0, stream>>>(s2, s2max);
  // 4) per-row softmax normalizers
  z_kernel<<<N, 256, 0, stream>>>(s1, s2, s2max, mrow, invZ);
  // 5) Wh^T (bf16) via WMMA
  wh_kernel<<<(NA / 16) * (OUTF / 16) / 8, 256, 0, stream>>>(Xbf, WTbf, whT);
  // 6) fused softmax @ Wh + ELU via WMMA
  attn_kernel<<<N / 64, 512, 0, stream>>>(s1, s2, mrow, invZ, whT, out);
}
